// PointNetAutoEncoder_65927747994190
// MI455X (gfx1250) — matrix operand order repaired and checked by reading, not compiled, not run on hardware
//
#include <hip/hip_runtime.h>
#include <hip/hip_bf16.h>

typedef __attribute__((ext_vector_type(16))) _Float16 v16h;
typedef __attribute__((ext_vector_type(8)))  _Float16 v8h;
typedef __attribute__((ext_vector_type(8)))  float    v8f;

#define LN_EPS 1e-3f

// =====================================================================
// WMMA GEMM: C[M,N] = act(A[M,K] @ W[K,N] + bias[N]);  act: 0=id, 1=relu
//   A : f16 row-major [M,K], K % 32 == 0 (f16 shadow of activations)
//   Wt: f16 TRANSPOSED [N,K] (pre-converted by prep_w_kernel)
//   C : f32 out; C16: optional f16 shadow out (for GEMM->GEMM chains)
//   One wave computes a 16 x (16*NT) tile: A fragment reused NT times.
//   K-loop is software-pipelined (double-buffered fragments, last
//   iteration peeled so the rotation can be register-renamed away).
//   Fragment packing: contiguous k = 16*(lane/16) + s in both A and B (16x32 A, 32x16 B).
// =====================================================================
template <int NT>
__global__ void gemm_wmma_kernel(const _Float16* __restrict__ A,
                                 const _Float16* __restrict__ Wt,
                                 const float* __restrict__ bias,
                                 float* __restrict__ C,
                                 _Float16* __restrict__ C16,
                                 int M, int N, int K, int act)
{
  const int lane   = threadIdx.x & 31;
  const int wave   = threadIdx.x >> 5;
  const int tilesN = N / (16 * NT);
  const int tile   = blockIdx.x * (blockDim.x >> 5) + wave;
  if (tile >= (M >> 4) * tilesN) return;      // wave-uniform: EXEC all-ones
  const int tm = (tile / tilesN) << 4;
  const int tn = (tile % tilesN) * (16 * NT);

  const int l15    = lane & 15;
  const int hi     = lane >> 4;               // 0 or 1
  const int kApart = hi * 16;                 // A: high lanes hold K=16..31
  const int kBpart = hi * 16;                 // B: high lanes hold K=16..31

  const v8h* Af = (const v8h*)(A + (size_t)(tm + l15) * K);
  const v8h* Bf[NT];
#pragma unroll
  for (int t = 0; t < NT; ++t)
    Bf[t] = (const v8h*)(Wt + (size_t)(tn + t * 16 + l15) * K);

  v8f acc[NT];
#pragma unroll
  for (int t = 0; t < NT; ++t) acc[t] = 0.0f;

  const int steps = K >> 5;

  // current fragments
  v8h a0, a1, b0[NT], b1[NT];

  auto burst = [&]() {
    v16h a;
#pragma unroll
    for (int j = 0; j < 8; ++j) { a[j] = a0[j]; a[8 + j] = a1[j]; }
#pragma unroll
    for (int t = 0; t < NT; ++t) {
      v16h b;
#pragma unroll
      for (int j = 0; j < 8; ++j) { b[j] = b0[t][j]; b[8 + j] = b1[t][j]; }
      acc[t] = __builtin_amdgcn_wmma_f32_16x16x32_f16(false, a, false, b,
                                                      (short)0, acc[t], false, false);
    }
  };

  // ---- prologue: load step 0 ----
  {
    const int ka = kApart >> 3;
    const int kb = kBpart >> 3;
    a0 = Af[ka]; a1 = Af[ka + 1];
#pragma unroll
    for (int t = 0; t < NT; ++t) { b0[t] = Bf[t][kb]; b1[t] = Bf[t][kb + 1]; }
  }

  // ---- steady state: load s+1 under the WMMA burst of s ----
#pragma unroll 2
  for (int s = 0; s + 1 < steps; ++s) {
    const int k0n = (s + 1) << 5;
    const int ka = (k0n + kApart) >> 3;
    const int kb = (k0n + kBpart) >> 3;
    __builtin_prefetch(Af + ka + 4, 0, 3);    // step s+2 into GL2
    v8h na0 = Af[ka], na1 = Af[ka + 1];
    v8h nb0[NT], nb1[NT];
#pragma unroll
    for (int t = 0; t < NT; ++t) { nb0[t] = Bf[t][kb]; nb1[t] = Bf[t][kb + 1]; }

    burst();

    a0 = na0; a1 = na1;
#pragma unroll
    for (int t = 0; t < NT; ++t) { b0[t] = nb0[t]; b1[t] = nb1[t]; }
  }

  // ---- epilogue: last WMMA burst ----
  burst();

  const int rbase = tm + hi * 8;
#pragma unroll
  for (int t = 0; t < NT; ++t) {
    const int col = tn + t * 16 + l15;
    const float bv = bias[col];
#pragma unroll
    for (int r = 0; r < 8; ++r) {
      float v = acc[t][r] + bv;
      if (act) v = fmaxf(v, 0.0f);
      const size_t idx = (size_t)(rbase + r) * N + col;
      C[idx] = v;
      if (C16) C16[idx] = (_Float16)v;
    }
  }
}

// =====================================================================
// Weight prep: Wt[n,k] = f16(W[k,n]), zero-padded to Kpad.
// =====================================================================
__global__ void prep_w_kernel(const float* __restrict__ W,
                              _Float16* __restrict__ Wt,
                              int K, int N, int Kpad)
{
  int i = blockIdx.x * blockDim.x + threadIdx.x;
  if (i >= N * Kpad) return;
  int n = i / Kpad, k = i % Kpad;
  Wt[i] = (_Float16)((k < K) ? W[(size_t)k * N + n] : 0.0f);
}

// Pad point cloud rows from 3 floats to 32 f16 (zero fill).
__global__ void pad_pc_kernel(const float* __restrict__ pc,
                              _Float16* __restrict__ out, int rows)
{
  int i = blockIdx.x * blockDim.x + threadIdx.x;
  if (i >= rows * 32) return;
  int r = i >> 5, k = i & 31;
  out[i] = (_Float16)((k < 3) ? pc[r * 3 + k] : 0.0f);
}

// pcr_pad16[r,bn,0:3] = f16(R[r] @ pc[bn]), rows padded to stride 32.
__global__ void rotate_pc_pad_kernel(const float* __restrict__ pc,
                                     const float* __restrict__ R,
                                     _Float16* __restrict__ pcr, int BN)
{
  int i = blockIdx.x * blockDim.x + threadIdx.x;
  if (i >= 4 * BN) return;
  int r = i / BN, bn = i % BN;
  const float* v  = pc + (size_t)bn * 3;
  const float* Rm = R + r * 9;
  _Float16* o = pcr + (size_t)i * 32;
#pragma unroll
  for (int ii = 0; ii < 3; ++ii)
    o[ii] = (_Float16)(Rm[ii * 3 + 0] * v[0] + Rm[ii * 3 + 1] * v[1] + Rm[ii * 3 + 2] * v[2]);
#pragma unroll
  for (int k = 3; k < 32; ++k) o[k] = (_Float16)0.0f;
}

// =====================================================================
// LayerNorm (per row over N) + ReLU, in place (+ optional f16 shadow).
// One wave32 per row.
// =====================================================================
__global__ void ln_relu_kernel(float* __restrict__ X,
                               _Float16* __restrict__ X16,
                               const float* __restrict__ g,
                               const float* __restrict__ b,
                               int M, int N)
{
  const int lane = threadIdx.x & 31;
  const int wave = threadIdx.x >> 5;
  const int row  = blockIdx.x * (blockDim.x >> 5) + wave;
  if (row >= M) return;
  float* x = X + (size_t)row * N;
  _Float16* x16 = X16 ? X16 + (size_t)row * N : (_Float16*)0;

  float s = 0.0f, s2 = 0.0f;
  for (int c = lane; c < N; c += 32) { float v = x[c]; s += v; s2 += v * v; }
#pragma unroll
  for (int off = 16; off > 0; off >>= 1) {
    s  += __shfl_xor(s,  off, 32);
    s2 += __shfl_xor(s2, off, 32);
  }
  const float m   = s / (float)N;
  const float var = s2 / (float)N - m * m;
  const float inv = rsqrtf(var + LN_EPS);
  for (int c = lane; c < N; c += 32) {
    float v = fmaxf(g[c] * (x[c] - m) * inv + b[c], 0.0f);
    x[c] = v;
    if (X16) x16[c] = (_Float16)v;
  }
}

// =====================================================================
// BatchNorm stats: per column c over M rows -> mean[c], invstd[c]
// =====================================================================
__global__ void bn_stats_kernel(const float* __restrict__ X,
                                float* __restrict__ mean,
                                float* __restrict__ invstd,
                                int M, int N)
{
  __shared__ float ss[256], ss2[256];
  const int c = blockIdx.x;
  float s = 0.0f, s2 = 0.0f;
  for (int r = threadIdx.x; r < M; r += blockDim.x) {
    float v = X[(size_t)r * N + c]; s += v; s2 += v * v;
  }
  ss[threadIdx.x] = s; ss2[threadIdx.x] = s2;
  __syncthreads();
  for (int st = 128; st > 0; st >>= 1) {
    if ((int)threadIdx.x < st) {
      ss[threadIdx.x]  += ss[threadIdx.x + st];
      ss2[threadIdx.x] += ss2[threadIdx.x + st];
    }
    __syncthreads();
  }
  if (threadIdx.x == 0) {
    float m = ss[0] / (float)M;
    float v = ss2[0] / (float)M - m * m;
    mean[c]   = m;
    invstd[c] = rsqrtf(v + LN_EPS);
  }
}

__global__ void bn_apply_kernel(float* __restrict__ X,
                                _Float16* __restrict__ X16,
                                const float* __restrict__ mean,
                                const float* __restrict__ invstd,
                                const float* __restrict__ g,
                                const float* __restrict__ b,
                                int M, int N)
{
  size_t i = (size_t)blockIdx.x * blockDim.x + threadIdx.x;
  if (i >= (size_t)M * N) return;
  int c = (int)(i % N);
  float v = g[c] * (X[i] - mean[c]) * invstd[c] + b[c];
  X[i] = v;
  if (X16) X16[i] = (_Float16)v;
}

// =====================================================================
// Max-pool over points: X[B,Np,F] -> out16[B,F] (feeds GEMMs only)
// =====================================================================
__global__ void maxpool_kernel(const float* __restrict__ X,
                               _Float16* __restrict__ out16,
                               int B, int Np, int F)
{
  int i = blockIdx.x * blockDim.x + threadIdx.x;
  if (i >= B * F) return;
  int b = i / F, f = i % F;
  const float* p = X + (size_t)b * Np * F + f;
  float m = -3.4e38f;
  for (int n = 0; n < Np; ++n) m = fmaxf(m, p[(size_t)n * F]);
  out16[i] = (_Float16)m;
}

// =====================================================================
// Small dense: C[16,4] = A[16,K] @ W[K,4] + b   (64 threads)
// =====================================================================
__global__ void dense16x4_kernel(const float* __restrict__ A,
                                 const float* __restrict__ W,
                                 const float* __restrict__ b,
                                 float* __restrict__ C, int K)
{
  int m = threadIdx.x >> 2, n = threadIdx.x & 3;
  float s = b[n];
  for (int k = 0; k < K; ++k) s += A[m * K + k] * W[k * 4 + n];
  C[m * 4 + n] = s;
}

__global__ void qnorm_kernel(const float* __restrict__ t, float* __restrict__ q)
{
  int b = threadIdx.x; // 16 threads
  float x = t[b * 4], y = t[b * 4 + 1], z = t[b * 4 + 2], w = t[b * 4 + 3];
  float inv = rsqrtf(x * x + y * y + z * z + w * w);
  q[b * 4] = x * inv; q[b * 4 + 1] = y * inv; q[b * 4 + 2] = z * inv; q[b * 4 + 3] = w * inv;
}

// =====================================================================
// rotated = v + w*t + cross(q.xyz, t), t = 2*cross(q.xyz, v); per point
// =====================================================================
__global__ void quat_transform_kernel(const float* __restrict__ dec,
                                      const float* __restrict__ q,
                                      float* __restrict__ out,
                                      int Np)   // per rotation launch, B=16
{
  int i = blockIdx.x * blockDim.x + threadIdx.x;
  if (i >= 16 * Np) return;
  int b = i / Np;
  const float* v = dec + (size_t)i * 3;
  float qx = q[b * 4], qy = q[b * 4 + 1], qz = q[b * 4 + 2], qw = q[b * 4 + 3];
  float tx = 2.0f * (qy * v[2] - qz * v[1]);
  float ty = 2.0f * (qz * v[0] - qx * v[2]);
  float tz = 2.0f * (qx * v[1] - qy * v[0]);
  out[(size_t)i * 3 + 0] = v[0] + qw * tx + (qy * tz - qz * ty);
  out[(size_t)i * 3 + 1] = v[1] + qw * ty + (qz * tx - qx * tz);
  out[(size_t)i * 3 + 2] = v[2] + qw * tz + (qx * ty - qy * tx);
}

// =====================================================================
// crc_loss = mean over 6 pairs x 16 batches of sum_f (enc_i - enc_j)^2
// =====================================================================
__global__ void crc_loss_kernel(const float* __restrict__ enc, float* __restrict__ out)
{
  __shared__ float sh[256];
  const int pi[6] = {0, 0, 0, 1, 1, 2};
  const int pj[6] = {1, 2, 3, 2, 3, 3};
  float acc = 0.0f;
  for (int idx = threadIdx.x; idx < 6 * 16 * 128; idx += blockDim.x) {
    int p = idx / (16 * 128);
    int rem = idx % (16 * 128);
    float d = enc[pi[p] * 2048 + rem] - enc[pj[p] * 2048 + rem];
    acc += d * d;
  }
  sh[threadIdx.x] = acc;
  __syncthreads();
  for (int st = 128; st > 0; st >>= 1) {
    if ((int)threadIdx.x < st) sh[threadIdx.x] += sh[threadIdx.x + st];
    __syncthreads();
  }
  if (threadIdx.x == 0) out[0] = sh[0] / 96.0f;
}

// =====================================================================
// overlap loss: per (r,b) cloud of 1024 pts; min sq-dist capped at 0.5.
// Points staged to LDS via GLOBAL_LOAD_ASYNC_TO_LDS_B128 (ASYNCcnt path).
// =====================================================================
__global__ void overlap_kernel(const float* __restrict__ rotated, float* __restrict__ out)
{
  __shared__ float pts[3072];                 // 1024 x (x,y,z) = 12288 B
  __shared__ float sh[256];
  const float* base = rotated + (size_t)blockIdx.x * 3072;

#if defined(__AMDGCN__)
  // Async DMA: 768 x 16B chunks, no VGPR round-trip, tracked by ASYNCcnt.
  {
    unsigned lbase = (unsigned)(size_t)&pts[0];   // LDS byte offset (flat low bits)
    for (int i = threadIdx.x; i < 768; i += blockDim.x) {
      unsigned long long ga = (unsigned long long)(size_t)base + (unsigned long long)i * 16u;
      unsigned lo = lbase + (unsigned)i * 16u;
      asm volatile("global_load_async_to_lds_b128 %0, %1, off"
                   :: "v"(lo), "v"(ga) : "memory");
    }
    asm volatile("s_wait_asynccnt 0" ::: "memory");
  }
#else
  for (int n = threadIdx.x; n < 3072; n += blockDim.x) pts[n] = base[n];
#endif
  __syncthreads();

  float acc = 0.0f;
  for (int n = threadIdx.x; n < 1024; n += blockDim.x) {
    float minv = 0.5f;                       // diagonal candidate = 2*RADIUS
    float x = pts[n * 3], y = pts[n * 3 + 1], z = pts[n * 3 + 2];
    for (int m = 0; m < 1024; ++m) {
      float dx = x - pts[m * 3], dy = y - pts[m * 3 + 1], dz = z - pts[m * 3 + 2];
      float d = dx * dx + dy * dy + dz * dz;
      d = (m == n) ? 0.5f : d;               // branchless diagonal
      minv = fminf(minv, d);
    }
    acc += (0.5f - minv);
  }
  sh[threadIdx.x] = acc;
  __syncthreads();
  for (int st = 128; st > 0; st >>= 1) {
    if ((int)threadIdx.x < st) sh[threadIdx.x] += sh[threadIdx.x + st];
    __syncthreads();
  }
  if (threadIdx.x == 0) atomicAdd(out, sh[0] * (1.0f / 65536.0f));
}

__global__ void zero_scalar_kernel(float* __restrict__ p) { p[0] = 0.0f; }

// =====================================================================
// Host-side orchestration
// =====================================================================
static void gemm(const _Float16* A, const _Float16* Wt, const float* b,
                 float* C, _Float16* C16, int M, int N, int K, int act, hipStream_t s)
{
  if (N % 64 == 0) {
    int tiles = (M / 16) * (N / 64);
    gemm_wmma_kernel<4><<<dim3((tiles + 7) / 8), dim3(256), 0, s>>>(A, Wt, b, C, C16, M, N, K, act);
  } else {
    int tiles = (M / 16) * (N / 32);
    gemm_wmma_kernel<2><<<dim3((tiles + 7) / 8), dim3(256), 0, s>>>(A, Wt, b, C, C16, M, N, K, act);
  }
}

static void ln_relu(float* X, _Float16* X16, const float* g, const float* b,
                    int M, int N, hipStream_t s)
{
  ln_relu_kernel<<<dim3((M + 7) / 8), dim3(256), 0, s>>>(X, X16, g, b, M, N);
}

static void bn(float* X, _Float16* X16, const float* g, const float* b,
               float* m, float* iv, int M, int N, hipStream_t s)
{
  bn_stats_kernel<<<dim3(N), dim3(256), 0, s>>>(X, m, iv, M, N);
  size_t total = (size_t)M * N;
  bn_apply_kernel<<<dim3((unsigned)((total + 255) / 256)), dim3(256), 0, s>>>(X, X16, m, iv, g, b, M, N);
}

struct WDesc { int idx, K, N, Kpad; };

extern "C" void kernel_launch(void* const* d_in, const int* in_sizes, int n_in,
                              void* d_out, int out_size, void* d_ws, size_t ws_size,
                              hipStream_t stream)
{
  const float* pc = (const float*)d_in[0];   // (16,1024,3)
  const float* R  = (const float*)d_in[1];   // (4,3,3)
  auto P = [&](int i) { return (const float*)d_in[i]; };

  const int B = 16, NP = 1024, MT = B * NP;  // 16384 transformer points

  float* ws = (float*)d_ws;
  size_t o = 0;
  float* bufA   = ws + o; o += (size_t)MT * 1024;   // big activation (64 MB)
  float* bufB   = ws + o; o += (size_t)MT * 128;
  float* bufC   = ws + o; o += (size_t)MT * 128;
  _Float16* bufB16 = (_Float16*)(ws + o); o += (size_t)MT * 64;
  _Float16* bufC16 = (_Float16*)(ws + o); o += (size_t)MT * 64;
  _Float16* pc16   = (_Float16*)(ws + o); o += (size_t)MT * 16;
  _Float16* pcr16  = (_Float16*)(ws + o); o += (size_t)4 * MT * 16;
  _Float16* pool16 = (_Float16*)(ws + o); o += 16 * 512;
  float* s1     = ws + o; o += 16 * 512;
  float* s2     = ws + o; o += 16 * 512;
  float* s3     = ws + o; o += 16 * 512;
  _Float16* s1h = (_Float16*)(ws + o); o += 16 * 256;
  _Float16* s2h = (_Float16*)(ws + o); o += 16 * 256;
  _Float16* s3h = (_Float16*)(ws + o); o += 16 * 256;
  float* tfm    = ws + o; o += 64;
  float* q      = ws + o; o += 64;
  float* enc    = ws + o; o += 4 * 16 * 128;
  _Float16* ench = (_Float16*)(ws + o); o += 4 * 16 * 64;
  float* dec    = ws + o; o += 16 * 3072;
  float* bnm    = ws + o; o += 1024;
  float* bniv   = ws + o; o += 1024;
  _Float16* wpool = (_Float16*)(ws + o); o += 1 << 20;   // 2M halfs

  float* rotated = (float*)d_out;            // 4*16*1024*3
  float* crc     = rotated + 196608;
  float* ovl     = rotated + 196609;

  // ---------- Prep: transpose+convert all GEMM weights to f16 [N,K] ----------
  static const WDesc wd[18] = {
    {2, 3, 64, 32},   {6, 64, 64, 64},   {10, 64, 64, 64},  {14, 64, 128, 64},
    {18, 128, 1024, 128}, {22, 1024, 512, 1024}, {26, 512, 256, 512},
    {32, 3, 32, 32},  {36, 32, 64, 32},  {40, 64, 64, 64},  {44, 64, 512, 64},
    {48, 512, 256, 512}, {52, 256, 128, 256}, {56, 128, 128, 128},
    {58, 128, 32, 128}, {62, 32, 64, 32}, {66, 64, 128, 64}, {70, 128, 3072, 128},
  };
  _Float16* wt[18];
  {
    size_t hoff = 0;
    for (int i = 0; i < 18; ++i) {
      wt[i] = wpool + hoff;
      int total = wd[i].N * wd[i].Kpad;
      hoff += (size_t)total;
      prep_w_kernel<<<dim3((total + 255) / 256), dim3(256), 0, stream>>>(
          P(wd[i].idx), wt[i], wd[i].K, wd[i].N, wd[i].Kpad);
    }
  }
  pad_pc_kernel<<<dim3((MT * 32 + 255) / 256), dim3(256), 0, stream>>>(pc, pc16, MT);
  rotate_pc_pad_kernel<<<dim3((4 * MT + 255) / 256), dim3(256), 0, stream>>>(pc, R, pcr16, MT);

  // ---------- Transformer: pointwise MLP + LN + ReLU ----------
  gemm(pc16,   wt[0], P(3),  bufB, 0, MT,   64,   32, 0, stream); ln_relu(bufB, bufB16, P(4),  P(5),  MT, 64,   stream);
  gemm(bufB16, wt[1], P(7),  bufC, 0, MT,   64,   64, 0, stream); ln_relu(bufC, bufC16, P(8),  P(9),  MT, 64,   stream);
  gemm(bufC16, wt[2], P(11), bufB, 0, MT,   64,   64, 0, stream); ln_relu(bufB, bufB16, P(12), P(13), MT, 64,   stream);
  gemm(bufB16, wt[3], P(15), bufC, 0, MT,  128,   64, 0, stream); ln_relu(bufC, bufC16, P(16), P(17), MT, 128,  stream);
  gemm(bufC16, wt[4], P(19), bufA, 0, MT, 1024,  128, 0, stream); ln_relu(bufA, 0,      P(20), P(21), MT, 1024, stream);
  maxpool_kernel<<<dim3((B * 1024 + 255) / 256), dim3(256), 0, stream>>>(bufA, pool16, B, NP, 1024);
  gemm(pool16, wt[5], P(23), s1, 0, 16, 512, 1024, 0, stream); ln_relu(s1, s1h, P(24), P(25), 16, 512, stream);
  gemm(s1h,    wt[6], P(27), s2, 0, 16, 256,  512, 0, stream); ln_relu(s2, s2h, P(28), P(29), 16, 256, stream);
  dense16x4_kernel<<<dim3(1), dim3(64), 0, stream>>>(s2, P(30), P(31), tfm, 256);
  qnorm_kernel<<<dim3(1), dim3(16), 0, stream>>>(tfm, q);

  // ---------- Encoder per rotation: ReLU then BatchNorm ----------
  for (int r = 0; r < 4; ++r) {
    const _Float16* Ar = pcr16 + (size_t)r * MT * 32;
    gemm(Ar,     wt[7],  P(33), bufB, 0, MT,  32, 32, 1, stream); bn(bufB, bufB16, P(34), P(35), bnm, bniv, MT, 32,  stream);
    gemm(bufB16, wt[8],  P(37), bufC, 0, MT,  64, 32, 1, stream); bn(bufC, bufC16, P(38), P(39), bnm, bniv, MT, 64,  stream);
    gemm(bufC16, wt[9],  P(41), bufB, 0, MT,  64, 64, 1, stream); bn(bufB, bufB16, P(42), P(43), bnm, bniv, MT, 64,  stream);
    gemm(bufB16, wt[10], P(45), bufA, 0, MT, 512, 64, 1, stream); bn(bufA, 0,      P(46), P(47), bnm, bniv, MT, 512, stream);
    maxpool_kernel<<<dim3((B * 512 + 255) / 256), dim3(256), 0, stream>>>(bufA, pool16, B, NP, 512);
    gemm(pool16, wt[11], P(49), s1, 0, 16, 256, 512, 1, stream); bn(s1, s1h, P(50), P(51), bnm, bniv, 16, 256, stream);
    gemm(s1h,    wt[12], P(53), s2, 0, 16, 128, 256, 1, stream); bn(s2, s2h, P(54), P(55), bnm, bniv, 16, 128, stream);
    gemm(s2h,    wt[13], P(57), enc + (size_t)r * 2048, ench + (size_t)r * 2048,
         16, 128, 128, 0, stream);
  }

  // ---------- Decoder per rotation + quaternion transform ----------
  for (int r = 0; r < 4; ++r) {
    const _Float16* zr = ench + (size_t)r * 2048;
    gemm(zr,  wt[14], P(59), s1, 0, 16,  32, 128, 1, stream); bn(s1, s1h, P(60), P(61), bnm, bniv, 16, 32,  stream);
    gemm(s1h, wt[15], P(63), s2, 0, 16,  64,  32, 1, stream); bn(s2, s2h, P(64), P(65), bnm, bniv, 16, 64,  stream);
    gemm(s2h, wt[16], P(67), s3, 0, 16, 128,  64, 1, stream); bn(s3, s3h, P(68), P(69), bnm, bniv, 16, 128, stream);
    gemm(s3h, wt[17], P(71), dec, 0, 16, 3072, 128, 0, stream);
    quat_transform_kernel<<<dim3((16 * NP + 255) / 256), dim3(256), 0, stream>>>(
        dec, q, rotated + (size_t)r * 16 * NP * 3, NP);
  }

  // ---------- Losses ----------
  crc_loss_kernel<<<dim3(1), dim3(256), 0, stream>>>(enc, crc);
  zero_scalar_kernel<<<dim3(1), dim3(1), 0, stream>>>(ovl);
  overlap_kernel<<<dim3(64), dim3(256), 0, stream>>>(rotated, ovl);

  (void)in_sizes; (void)n_in; (void)out_size; (void)ws_size;
}
